// BitLinearCuda_36825049596368
// MI455X (gfx1250) — 8-bit matrix operand order repaired and checked by reading, compiled, not run on hardware
//
#include <hip/hip_runtime.h>

#define M_TOT 16384
#define N_TOT 2048
#define K_TOT 2048

typedef __attribute__((ext_vector_type(8))) int v8i;
typedef __attribute__((ext_vector_type(4))) int v4i;
typedef __attribute__((ext_vector_type(2))) int v2i;

__global__ __launch_bounds__(256) void k_wabs_partial(const float* __restrict__ w,
                                                      float* __restrict__ partial) {
    const int tid = blockIdx.x * 256 + threadIdx.x;
    float s = 0.0f;
    #pragma unroll
    for (int i = 0; i < 16; ++i)
        s += fabsf(w[(size_t)i * (1024 * 256) + tid]);
    __shared__ float red[256];
    red[threadIdx.x] = s;
    __syncthreads();
    for (int st = 128; st > 0; st >>= 1) {
        if (threadIdx.x < st) red[threadIdx.x] += red[threadIdx.x + st];
        __syncthreads();
    }
    if (threadIdx.x == 0) partial[blockIdx.x] = red[0];
}

__global__ __launch_bounds__(256) void k_gamma(const float* __restrict__ partial,
                                               float* __restrict__ gma) {
    float s = 0.0f;
    #pragma unroll
    for (int i = 0; i < 4; ++i) s += partial[threadIdx.x + i * 256];
    __shared__ float red[256];
    red[threadIdx.x] = s;
    __syncthreads();
    for (int st = 128; st > 0; st >>= 1) {
        if (threadIdx.x < st) red[threadIdx.x] += red[threadIdx.x + st];
        __syncthreads();
    }
    if (threadIdx.x == 0) {
        float g = fmaxf(red[0] * (1.0f / 4194304.0f), 1e-5f);
        gma[0] = g;
        gma[1] = 1.0f / g;
    }
}

__global__ __launch_bounds__(256) void k_quant_w(const float* __restrict__ w,
                                                 signed char* __restrict__ wq,
                                                 const float* __restrict__ gma) {
    const float invg = gma[1];
    const int t = blockIdx.x * 256 + threadIdx.x;
    float4 v = ((const float4*)w)[t];
    auto q = [&](float f) -> int {
        float c = fminf(fmaxf(f * invg, -1.0f), 1.0f);
        return ((int)rintf(c)) & 0xff;
    };
    ((int*)wq)[t] = q(v.x) | (q(v.y) << 8) | (q(v.z) << 16) | (q(v.w) << 24);
}

__global__ __launch_bounds__(256) void k_quant_act(const float* __restrict__ x,
                                                   signed char* __restrict__ xq,
                                                   float* __restrict__ xscale) {
    const int row = blockIdx.x;
    const int t = threadIdx.x;
    const float4* xr = (const float4*)(x + (size_t)row * K_TOT);
    float4 a = xr[t * 2 + 0];
    float4 b = xr[t * 2 + 1];
    float m = fmaxf(fmaxf(fmaxf(fabsf(a.x), fabsf(a.y)), fmaxf(fabsf(a.z), fabsf(a.w))),
                    fmaxf(fmaxf(fabsf(b.x), fabsf(b.y)), fmaxf(fabsf(b.z), fabsf(b.w))));
    __shared__ float red[256];
    red[t] = m;
    __syncthreads();
    for (int st = 128; st > 0; st >>= 1) {
        if (t < st) red[t] = fmaxf(red[t], red[t + st]);
        __syncthreads();
    }
    const float eta = fmaxf(red[0], 1e-5f);
    const float inv = 127.0f / eta;
    if (t == 0) xscale[row] = eta * (1.0f / 127.0f);
    auto q8 = [&](float v) -> int {
        float q = rintf(v * inv);
        q = fminf(fmaxf(q, -128.0f), 127.0f);
        return ((int)q) & 0xff;
    };
    int lo = q8(a.x) | (q8(a.y) << 8) | (q8(a.z) << 16) | (q8(a.w) << 24);
    int hi = q8(b.x) | (q8(b.y) << 8) | (q8(b.z) << 16) | (q8(b.w) << 24);
    ((int2*)(xq + (size_t)row * K_TOT))[t] = make_int2(lo, hi);
}

__global__ __launch_bounds__(256) void k_gemm_wmma(
    const signed char* __restrict__ xq,
    const signed char* __restrict__ wq,
    const float* __restrict__ xscale,
    const float* __restrict__ gma,
    const float* __restrict__ bias,
    float* __restrict__ out) {

    const int lane   = threadIdx.x & 31;
    const int wave   = threadIdx.x >> 5;
    const int half   = lane >> 4;
    const int laneLo = lane & 15;

    const int waveM = wave & 3;
    const int waveN = wave >> 2;
    const int mBase = blockIdx.y * 64 + waveM * 16;
    const int nBase = blockIdx.x * 128 + waveN * 64;

    const signed char* aRow  = xq + (size_t)(mBase + laneLo) * K_TOT + half * 16;
    const signed char* bRow0 = wq + (size_t)(nBase + laneLo) * K_TOT + half * 16;

    v8i acc[4] = {v8i{}, v8i{}, v8i{}, v8i{}};

    #pragma unroll 2
    for (int k = 0; k < K_TOT; k += 64) {
        v8i a;
        const v2i a0 = *(const v2i*)(aRow + k + 0);
        const v2i a1 = *(const v2i*)(aRow + k + 8);
        const v2i a2 = *(const v2i*)(aRow + k + 32);
        const v2i a3 = *(const v2i*)(aRow + k + 40);
        a[0] = a0[0]; a[1] = a0[1];
        a[2] = a1[0]; a[3] = a1[1];
        a[4] = a2[0]; a[5] = a2[1];
        a[6] = a3[0]; a[7] = a3[1];

        __builtin_prefetch(aRow + k + 64, 0, 1);

        #pragma unroll
        for (int tc = 0; tc < 4; ++tc) {
            const signed char* bRow = bRow0 + (size_t)tc * 16 * K_TOT;
            const v4i b0 = *(const v4i*)(bRow + k);
            const v4i b1 = *(const v4i*)(bRow + k + 32);
            v8i b;
            b[0] = b0[0]; b[1] = b0[1]; b[2] = b0[2]; b[3] = b0[3];
            b[4] = b1[0]; b[5] = b1[1]; b[6] = b1[2]; b[7] = b1[3];
            acc[tc] = __builtin_amdgcn_wmma_i32_16x16x64_iu8(
                true, a, true, b, acc[tc], false, false);
        }
    }

    const float g = gma[0];
    float rowScale[8];
    #pragma unroll
    for (int r = 0; r < 8; ++r)
        rowScale[r] = xscale[mBase + half * 8 + r] * g;

    #pragma unroll
    for (int tc = 0; tc < 4; ++tc) {
        const int n = nBase + tc * 16 + laneLo;
        const float bn = bias[n];
        #pragma unroll
        for (int r = 0; r < 8; ++r) {
            const int m = mBase + half * 8 + r;
            out[(size_t)m * N_TOT + n] = (float)acc[tc][r] * rowScale[r] + bn;
        }
    }
}

extern "C" void kernel_launch(void* const* d_in, const int* in_sizes, int n_in,
                              void* d_out, int out_size, void* d_ws, size_t ws_size,
                              hipStream_t stream) {
    (void)in_sizes; (void)n_in; (void)out_size; (void)ws_size;

    const float* x    = (const float*)d_in[0];
    const float* w    = (const float*)d_in[1];
    const float* bias = (const float*)d_in[2];
    float* out        = (float*)d_out;

    char* ws = (char*)d_ws;
    signed char* xq      = (signed char*)ws;
    signed char* wq      = (signed char*)(ws + (size_t)33554432);
    float*       xscale  = (float*)(ws + (size_t)37748736);
    float*       partial = (float*)(ws + (size_t)37814272);
    float*       gma     = (float*)(ws + (size_t)37818368);

    k_wabs_partial<<<1024, 256, 0, stream>>>(w, partial);
    k_gamma<<<1, 256, 0, stream>>>(partial, gma);
    k_quant_w<<<4096, 256, 0, stream>>>(w, wq, gma);
    k_quant_act<<<M_TOT, 256, 0, stream>>>(x, xq, xscale);

    dim3 grid(N_TOT / 128, M_TOT / 64);
    k_gemm_wmma<<<grid, 256, 0, stream>>>(xq, wq, xscale, gma, bias, out);
}
